// SEGNN_77000173683168
// MI455X (gfx1250) — matrix operand order repaired and checked by reading, not compiled, not run on hardware
//
#include <hip/hip_runtime.h>

// ---------------------------------------------------------------------------
// SEGNN forward for MI455X (gfx1250).
// All tensor-product stages are lowered to bf16 WMMA GEMMs
// (v_wmma_f32_16x16x32_bf16, f32 accumulate). One wave32 owns a 16-row tile.
// Fragments are assembled from LDS with the documented CDNA5 VGPR layouts.
// ---------------------------------------------------------------------------

#define N_NODES  25000
#define N_EDGES  400000
#define N_GRAPHS 512

#define INV3F 0.5773502691896258f
#define INV2F 0.7071067811865475f

typedef __bf16 bf16_t;
typedef bf16_t v16bf __attribute__((ext_vector_type(16)));
typedef float  v8f   __attribute__((ext_vector_type(8)));

union FragAB { v16bf v; unsigned u[8]; };

__device__ __forceinline__ unsigned pack2bf(float lo, float hi) {
  unsigned a = __float_as_uint(lo);
  unsigned b = __float_as_uint(hi);
  a = (a + 0x7FFFu + ((a >> 16) & 1u)) >> 16;   // RNE f32 -> bf16
  b = (b + 0x7FFFu + ((b >> 16) & 1u)) >> 16;
  return (a & 0xFFFFu) | (b << 16);
}
__device__ __forceinline__ float silu_f(float x){ return x / (1.f + __expf(-x)); }
__device__ __forceinline__ float sigm_f(float x){ return 1.f / (1.f + __expf(-x)); }

// ---------------------------------------------------------------------------
// Wave-level GEMM: D[16][M] = A[16][2*KP] * W[2*KP][M], bf16 inputs, f32 acc.
// Apk: LDS, K-pair-interleaved [KP][16] uints (pair of consecutive K per uint)
// Wpk: LDS, K-pair-interleaved [KP][M] uints
// A-frag slot r: kpair = (r&3) + 4h + 8(r>>2)  (ISA 16-bit A 16x32 layout)
// B-frag slot r: kpair = (r&3) + 4h + 8(r>>2)  (same k order as A)
// D   slot r: row = r + 8h, col = lane&15
// ---------------------------------------------------------------------------
template<int KP, int M>
__device__ __forceinline__ void wave_gemm(const unsigned* Apk, const unsigned* Wpk,
                                          float* out, int out_stride, int lane)
{
  const int n = lane & 15;
  const int h = lane >> 4;
  for (int nt = 0; nt < M/16; ++nt) {
    v8f acc = {0.f,0.f,0.f,0.f,0.f,0.f,0.f,0.f};
    for (int kb = 0; kb < KP/16; ++kb) {
      FragAB a, b;
#pragma unroll
      for (int r = 0; r < 8; ++r) {
        const int kpa = (kb<<4) + (r & 3) + (h << 2) + ((r >> 2) << 3);
        a.u[r] = Apk[(kpa << 4) + n];
        const int kpb = (kb<<4) + (r & 3) + (h << 2) + ((r >> 2) << 3);
        b.u[r] = Wpk[kpb * M + (nt << 4) + n];
      }
      acc = __builtin_amdgcn_wmma_f32_16x16x32_bf16(false, a.v, false, b.v,
                                                    (short)0, acc, false, false);
    }
#pragma unroll
    for (int r = 0; r < 8; ++r)
      out[(r + (h << 3)) * out_stride + (nt << 4) + n] = acc[r];
  }
}

// Per-wave LDS scratch (fixed strides; widths used depend on stage dims)
struct WaveBufs {
  float* feat;    // [16][128]  scalar features
  float* vfeat;   // [3][16][64] vector features (component major)
  float* ubuf;    // [16][64]   u_i = v_i . a_v
  float* raws;    // [16][96]   raw scalar GEMM output
  float* outv;    // [3][16][32] vector GEMM output
  float* as_;     // [16]
  float* av;      // [16][3]
  int*   edst;    // [16]
  int*   esrc;    // [16]
  unsigned* apk;  // [128][16]  packed bf16 A (pair-interleaved)
};

#define PW_BYTES (16*128*4 + 3*16*64*4 + 16*64*4 + 16*96*4 + 3*16*32*4 \
                  + 16*4 + 16*3*4 + 16*4 + 16*4 + 128*16*4)   /* 45440 */

__device__ __forceinline__ WaveBufs carve_wave(unsigned char* p) {
  WaveBufs B;
  B.feat  = (float*)p;     p += 16*128*4;
  B.vfeat = (float*)p;     p += 3*16*64*4;
  B.ubuf  = (float*)p;     p += 16*64*4;
  B.raws  = (float*)p;     p += 16*96*4;
  B.outv  = (float*)p;     p += 3*16*32*4;
  B.as_   = (float*)p;     p += 16*4;
  B.av    = (float*)p;     p += 16*3*4;
  B.edst  = (int*)p;       p += 16*4;
  B.esrc  = (int*)p;       p += 16*4;
  B.apk   = (unsigned*)p;
  return B;
}

// ---------------------------------------------------------------------------
// Generic tensor product (optionally gated) for one 16-row tile.
//   out_s = [s*a_s , u] @ Ws (+bias)          (u = v.a_v; INV3 folded in Ws)
//   out_v[c] = [s*a_v_c , v_c*a_s , cross(v,a_v)_c] @ Wv   (INV2 folded in Wv)
// GATE: silu on first M0-M1 scalars, sigmoid-gate the M1 vector channels.
// ---------------------------------------------------------------------------
template<int N0I, int N1I, int M0, int M1, bool GATE>
__device__ void tp_wave(const WaveBufs& B,
                        const unsigned* Ws, const unsigned* Wv, const float* bias,
                        float* dst_s, int dss, float* dst_v, int dvc, int dvs,
                        int lane)
{
  constexpr int KS = N0I + N1I;
  constexpr int KV = N0I + 2*N1I;

  // u[e][i] = sum_c v[c][e][i] * a_v[e][c]
  for (int i = lane; i < 16*N1I; i += 32) {
    int e = i / N1I, k = i - e*N1I;
    B.ubuf[e*64 + k] =
        B.vfeat[(0*16+e)*64 + k]*B.av[e*3+0] +
        B.vfeat[(1*16+e)*64 + k]*B.av[e*3+1] +
        B.vfeat[(2*16+e)*64 + k]*B.av[e*3+2];
  }
  // scalar A (bf16, pair-interleaved [KS/2][16])
  for (int i = lane; i < (KS/2)*16; i += 32) {
    int kp = i >> 4, e = i & 15;
    int k0 = kp*2, k1 = k0 + 1;
    float x0 = (k0 < N0I) ? B.feat[e*128+k0]*B.as_[e] : B.ubuf[e*64 + (k0-N0I)];
    float x1 = (k1 < N0I) ? B.feat[e*128+k1]*B.as_[e] : B.ubuf[e*64 + (k1-N0I)];
    B.apk[(kp<<4) + e] = pack2bf(x0, x1);
  }
  wave_gemm<KS/2, M0>(B.apk, Ws, B.raws, 96, lane);

  // vector components
  if (M1 > 0) {
    for (int c = 0; c < 3; ++c) {
      const int c1 = (c+1)%3, c2 = (c+2)%3;
      for (int i = lane; i < (KV/2)*16; i += 32) {
        int kp = i >> 4, e = i & 15;
        float xx[2];
#pragma unroll
        for (int t = 0; t < 2; ++t) {
          int k = kp*2 + t;
          float val;
          if (k < N0I)          val = B.feat[e*128+k]*B.av[e*3+c];
          else if (k < N0I+N1I) val = B.vfeat[(c*16+e)*64 + (k-N0I)]*B.as_[e];
          else {
            int kk = k - N0I - N1I;
            val = B.vfeat[(c1*16+e)*64+kk]*B.av[e*3+c2]
                - B.vfeat[(c2*16+e)*64+kk]*B.av[e*3+c1];
          }
          xx[t] = val;
        }
        B.apk[(kp<<4) + e] = pack2bf(xx[0], xx[1]);
      }
      wave_gemm<KV/2, M1>(B.apk, Wv, B.outv + c*16*32, 32, lane);
    }
  }

  // epilogue (loops uniform across the wave: widths are multiples of 32)
  if (GATE) {
    constexpr int MSo = M0 - M1;
    for (int e = 0; e < 16; ++e)
      for (int k = lane; k < MSo; k += 32)
        dst_s[e*dss + k] = silu_f(B.raws[e*96 + k] + bias[k]);
    for (int e = 0; e < 16; ++e)
      for (int k = lane; k < M1; k += 32) {
        float g = sigm_f(B.raws[e*96 + MSo + k] + bias[MSo + k]);
        dst_v[0*dvc + e*dvs + k] = B.outv[(0*16+e)*32 + k]*g;
        dst_v[1*dvc + e*dvs + k] = B.outv[(1*16+e)*32 + k]*g;
        dst_v[2*dvc + e*dvs + k] = B.outv[(2*16+e)*32 + k]*g;
      }
  } else {
    for (int e = 0; e < 16; ++e)
      for (int k = lane; k < M0; k += 32)
        dst_s[e*dss + k] = B.raws[e*96 + k] + bias[k];
    for (int e = 0; e < 16; ++e)
      for (int k = lane; k < M1; k += 32) {
        dst_v[0*dvc + e*dvs + k] = B.outv[(0*16+e)*32 + k];
        dst_v[1*dvc + e*dvs + k] = B.outv[(1*16+e)*32 + k];
        dst_v[2*dvc + e*dvs + k] = B.outv[(2*16+e)*32 + k];
      }
  }
}

// ------------------------- weight pre-packing ------------------------------
// dst[kp*m + col] = bf16pair( W(2kp,col), W(2kp+1,col) ), W = stacked blocks.
__global__ void pack_stacked(const float* w0, int n0, float s0,
                             const float* w1, int n1, float s1,
                             const float* w2, int n2, float s2,
                             int m, unsigned* dst)
{
  int K = n0 + n1 + n2;
  int total = (K >> 1) * m;
  for (int i = blockIdx.x*blockDim.x + threadIdx.x; i < total;
       i += gridDim.x*blockDim.x) {
    int kp = i / m, col = i - kp*m;
    float x[2];
#pragma unroll
    for (int t = 0; t < 2; ++t) {
      int k = kp*2 + t;
      float val;
      if (k < n0)           val = w0[k*m + col] * s0;
      else if (k < n0+n1)   val = w1[(k-n0)*m + col] * s1;
      else                  val = w2[(k-n0-n1)*m + col] * s2;
      x[t] = val;
    }
    dst[i] = pack2bf(x[0], x[1]);
  }
}

// ------------------------------ embedding ----------------------------------
__global__ void embed_kernel(const float* __restrict__ x,
                             const float* __restrict__ node_attr,
                             const float* __restrict__ w000, const float* __restrict__ w110,
                             const float* __restrict__ w011, const float* __restrict__ w101,
                             const float* __restrict__ w111, const float* __restrict__ bias,
                             float* __restrict__ s, float* __restrict__ v)
{
  int n = blockIdx.x*blockDim.x + threadIdx.x;
  if (n >= N_NODES) return;
  float si[8], vv[3], av[3];
#pragma unroll
  for (int i = 0; i < 8; ++i) si[i] = x[n*11 + i];
#pragma unroll
  for (int c = 0; c < 3; ++c) { vv[c] = x[n*11 + 8 + c]; av[c] = node_attr[n*4 + 1 + c]; }
  float u = vv[0]*av[0] + vv[1]*av[1] + vv[2]*av[2];
  float cr[3];
#pragma unroll
  for (int c = 0; c < 3; ++c) {
    int c1 = (c+1)%3, c2 = (c+2)%3;
    cr[c] = vv[c1]*av[c2] - vv[c2]*av[c1];
  }
  for (int k = 0; k < 64; ++k) {
    float o = bias[k] + u*INV3F*w110[k];
#pragma unroll
    for (int i = 0; i < 8; ++i) o += si[i]*w000[i*64 + k];
    s[n*64 + k] = o;
  }
  for (int k = 0; k < 32; ++k) {
    float A = 0.f;
#pragma unroll
    for (int i = 0; i < 8; ++i) A += si[i]*w011[i*32 + k];
#pragma unroll
    for (int c = 0; c < 3; ++c)
      v[(size_t)n*96 + c*32 + k] = A*av[c] + vv[c]*w101[k] + cr[c]*w111[k]*INV2F;
  }
}

// ------------------------------- messages ----------------------------------
#define MW1S 9216
#define MW1V 4096
#define MW2S 4608
#define MW2V 2048
#define SMEM_MSG ((MW1S+MW1V+MW2S+MW2V)*4 + 192*4 + 4*PW_BYTES)

__global__ __launch_bounds__(128) void message_kernel(
    const float* __restrict__ s, const float* __restrict__ v,
    const int* __restrict__ edge_index, const float* __restrict__ edge_attr,
    const unsigned* __restrict__ W1s, const unsigned* __restrict__ W1v,
    const float* __restrict__ bias1,
    const unsigned* __restrict__ W2s, const unsigned* __restrict__ W2v,
    const float* __restrict__ bias2,
    float* __restrict__ ag_s, float* __restrict__ ag_v)
{
  extern __shared__ unsigned char smem[];
  unsigned* ws1 = (unsigned*)smem;
  unsigned* wv1 = ws1 + MW1S;
  unsigned* ws2 = wv1 + MW1V;
  unsigned* wv2 = ws2 + MW2S;
  float* b1 = (float*)(wv2 + MW2V);
  float* b2 = b1 + 96;
  unsigned char* pw = (unsigned char*)(b2 + 96);

  const int t = threadIdx.x;
  for (int i = t; i < MW1S; i += 128) ws1[i] = W1s[i];
  for (int i = t; i < MW1V; i += 128) wv1[i] = W1v[i];
  for (int i = t; i < MW2S; i += 128) ws2[i] = W2s[i];
  for (int i = t; i < MW2V; i += 128) wv2[i] = W2v[i];
  for (int i = t; i < 96;   i += 128) { b1[i] = bias1[i]; b2[i] = bias2[i]; }
  __syncthreads();

  const int w = t >> 5, lane = t & 31;
  WaveBufs B = carve_wave(pw + (size_t)w * PW_BYTES);
  const int e0 = (blockIdx.x*4 + w) * 16;   // E divisible by 64

  if (lane < 16) {
    B.esrc[lane] = edge_index[e0 + lane];
    B.edst[lane] = edge_index[N_EDGES + e0 + lane];
    B.as_[lane]  = edge_attr[(size_t)(e0 + lane)*4 + 0];
  } else {
    int e = lane - 16;
    B.av[e*3+0] = edge_attr[(size_t)(e0+e)*4 + 1];
    B.av[e*3+1] = edge_attr[(size_t)(e0+e)*4 + 2];
    B.av[e*3+2] = edge_attr[(size_t)(e0+e)*4 + 3];
  }
  // gather: ms = [s[dst], s[src]], mv = [v[dst], v[src]]
  for (int i = lane; i < 16*64; i += 32) {
    int e = i >> 6, k = i & 63;
    int nd = B.edst[e], ns = B.esrc[e];
    B.feat[e*128 + k]      = s[(size_t)nd*64 + k];
    B.feat[e*128 + 64 + k] = s[(size_t)ns*64 + k];
  }
  for (int i = lane; i < 16*96; i += 32) {
    int e = i / 96, r = i - e*96, c = r >> 5, k = r & 31;
    int nd = B.edst[e], ns = B.esrc[e];
    B.vfeat[(c*16+e)*64 + k]      = v[(size_t)nd*96 + c*32 + k];
    B.vfeat[(c*16+e)*64 + 32 + k] = v[(size_t)ns*96 + c*32 + k];
  }

  // msg1 (gate) chained into msg2 (gate), all in LDS
  tp_wave<128,64,96,32,true>(B, ws1, wv1, b1, B.feat, 128, B.vfeat, 1024, 64, lane);
  tp_wave< 64,32,96,32,true>(B, ws2, wv2, b2, B.feat, 128, B.vfeat, 1024, 64, lane);

  // scatter-add (segment sum over dst)
  for (int i = lane; i < 16*64; i += 32) {
    int e = i >> 6, k = i & 63;
    atomicAdd(&ag_s[(size_t)B.edst[e]*64 + k], B.feat[e*128 + k]);
  }
  for (int i = lane; i < 16*96; i += 32) {
    int e = i / 96, r = i - e*96, c = r >> 5, k = r & 31;
    atomicAdd(&ag_v[(size_t)B.edst[e]*96 + c*32 + k], B.vfeat[(c*16+e)*64 + k]);
  }
}

// ------------------------------- node update -------------------------------
#define UW1S 9216
#define UW1V 4096
#define UW2S 3072
#define UW2V 2048
#define SMEM_UPD ((UW1S+UW1V+UW2S+UW2V)*4 + 160*4 + 4*PW_BYTES)

__global__ __launch_bounds__(128) void update_kernel(
    float* __restrict__ s, float* __restrict__ v,
    const float* __restrict__ ag_s, const float* __restrict__ ag_v,
    const float* __restrict__ node_attr,
    const unsigned* __restrict__ W1s, const unsigned* __restrict__ W1v,
    const float* __restrict__ bias1,
    const unsigned* __restrict__ W2s, const unsigned* __restrict__ W2v,
    const float* __restrict__ bias2)
{
  extern __shared__ unsigned char smem[];
  unsigned* ws1 = (unsigned*)smem;
  unsigned* wv1 = ws1 + UW1S;
  unsigned* ws2 = wv1 + UW1V;
  unsigned* wv2 = ws2 + UW2S;
  float* b1 = (float*)(wv2 + UW2V);
  float* b2 = b1 + 96;
  unsigned char* pw = (unsigned char*)(b2 + 64);

  const int t = threadIdx.x;
  for (int i = t; i < UW1S; i += 128) ws1[i] = W1s[i];
  for (int i = t; i < UW1V; i += 128) wv1[i] = W1v[i];
  for (int i = t; i < UW2S; i += 128) ws2[i] = W2s[i];
  for (int i = t; i < UW2V; i += 128) wv2[i] = W2v[i];
  for (int i = t; i < 96;   i += 128) b1[i] = bias1[i];
  for (int i = t; i < 64;   i += 128) b2[i] = bias2[i];
  __syncthreads();

  const int w = t >> 5, lane = t & 31;
  WaveBufs B = carve_wave(pw + (size_t)w * PW_BYTES);
  const int nd0 = (blockIdx.x*4 + w) * 16;

  if (lane < 16) {
    int nd = nd0 + lane;
    bool ok = nd < N_NODES;
    B.as_[lane] = 1.f;  // node a_s is forced to 1.0
    B.av[lane*3+0] = ok ? node_attr[(size_t)nd*4 + 1] : 0.f;
    B.av[lane*3+1] = ok ? node_attr[(size_t)nd*4 + 2] : 0.f;
    B.av[lane*3+2] = ok ? node_attr[(size_t)nd*4 + 3] : 0.f;
  }
  for (int i = lane; i < 16*64; i += 32) {
    int e = i >> 6, k = i & 63;
    int nd = nd0 + e; bool ok = nd < N_NODES;
    B.feat[e*128 + k]      = ok ? s[(size_t)nd*64 + k]    : 0.f;
    B.feat[e*128 + 64 + k] = ok ? ag_s[(size_t)nd*64 + k] : 0.f;
  }
  for (int i = lane; i < 16*96; i += 32) {
    int e = i / 96, r = i - e*96, c = r >> 5, k = r & 31;
    int nd = nd0 + e; bool ok = nd < N_NODES;
    B.vfeat[(c*16+e)*64 + k]      = ok ? v[(size_t)nd*96 + c*32 + k]    : 0.f;
    B.vfeat[(c*16+e)*64 + 32 + k] = ok ? ag_v[(size_t)nd*96 + c*32 + k] : 0.f;
  }

  tp_wave<128,64,96,32,true >(B, ws1, wv1, b1, B.feat, 128, B.vfeat, 1024, 64, lane);
  tp_wave< 64,32,64,32,false>(B, ws2, wv2, b2, B.raws,  96, B.outv,   512, 32, lane);

  // residual add
  for (int i = lane; i < 16*64; i += 32) {
    int e = i >> 6, k = i & 63;
    int nd = nd0 + e;
    if (nd < N_NODES) s[(size_t)nd*64 + k] += B.raws[e*96 + k];
  }
  for (int i = lane; i < 16*96; i += 32) {
    int e = i / 96, r = i - e*96, c = r >> 5, k = r & 31;
    int nd = nd0 + e;
    if (nd < N_NODES) v[(size_t)nd*96 + c*32 + k] += B.outv[(c*16+e)*32 + k];
  }
}

// ------------------------------ pre / readout ------------------------------
#define PW1S 4608
#define PW1V 2048
#define PW2S 4608
#define SMEM_PRE ((PW1S+PW1V+PW2S)*4 + 192*4 + 4*PW_BYTES)

__global__ __launch_bounds__(128) void pre_kernel(
    const float* __restrict__ s, const float* __restrict__ v,
    const float* __restrict__ node_attr,
    const unsigned* __restrict__ W1s, const unsigned* __restrict__ W1v,
    const float* __restrict__ bias1,
    const unsigned* __restrict__ W2s, const float* __restrict__ bias2,
    float* __restrict__ ps)
{
  extern __shared__ unsigned char smem[];
  unsigned* ws1 = (unsigned*)smem;
  unsigned* wv1 = ws1 + PW1S;
  unsigned* ws2 = wv1 + PW1V;
  float* b1 = (float*)(ws2 + PW2S);
  float* b2 = b1 + 96;
  unsigned char* pw = (unsigned char*)(b2 + 96);

  const int t = threadIdx.x;
  for (int i = t; i < PW1S; i += 128) ws1[i] = W1s[i];
  for (int i = t; i < PW1V; i += 128) wv1[i] = W1v[i];
  for (int i = t; i < PW2S; i += 128) ws2[i] = W2s[i];
  for (int i = t; i < 96;   i += 128) { b1[i] = bias1[i]; b2[i] = bias2[i]; }
  __syncthreads();

  const int w = t >> 5, lane = t & 31;
  WaveBufs B = carve_wave(pw + (size_t)w * PW_BYTES);
  const int nd0 = (blockIdx.x*4 + w) * 16;

  if (lane < 16) {
    int nd = nd0 + lane;
    bool ok = nd < N_NODES;
    B.as_[lane] = 1.f;
    B.av[lane*3+0] = ok ? node_attr[(size_t)nd*4 + 1] : 0.f;
    B.av[lane*3+1] = ok ? node_attr[(size_t)nd*4 + 2] : 0.f;
    B.av[lane*3+2] = ok ? node_attr[(size_t)nd*4 + 3] : 0.f;
  }
  for (int i = lane; i < 16*64; i += 32) {
    int e = i >> 6, k = i & 63;
    int nd = nd0 + e;
    B.feat[e*128 + k] = (nd < N_NODES) ? s[(size_t)nd*64 + k] : 0.f;
  }
  for (int i = lane; i < 16*96; i += 32) {
    int e = i / 96, r = i - e*96, c = r >> 5, k = r & 31;
    int nd = nd0 + e;
    B.vfeat[(c*16+e)*64 + k] = (nd < N_NODES) ? v[(size_t)nd*96 + c*32 + k] : 0.f;
  }

  tp_wave<64,32,96,32,true >(B, ws1, wv1, b1, B.feat, 128, B.vfeat, 1024, 64, lane);
  tp_wave<64,32,96, 0,false>(B, ws2, nullptr, b2, B.raws, 96, B.outv, 512, 32, lane);

  for (int i = lane; i < 16*96; i += 32) {
    int e = i / 96, k = i - e*96;
    int nd = nd0 + e;
    if (nd < N_NODES) ps[(size_t)nd*96 + k] = B.raws[e*96 + k];
  }
}

// ------------------------------- pooling / MLP -----------------------------
__global__ void zero_kernel(float* p, size_t n) {
  for (size_t i = blockIdx.x*(size_t)blockDim.x + threadIdx.x; i < n;
       i += (size_t)gridDim.x*blockDim.x) p[i] = 0.f;
}

__global__ void pool_kernel(const float* __restrict__ ps,
                            const int* __restrict__ batch,
                            float* __restrict__ g, float* __restrict__ cnt)
{
  int i = blockIdx.x*blockDim.x + threadIdx.x;
  if (i >= N_NODES*96) return;
  int n = i / 96, k = i - n*96;
  int b = batch[n];
  atomicAdd(&g[b*96 + k], ps[i]);
  if (k == 0) atomicAdd(&cnt[b], 1.f);
}

__global__ void mlp_kernel(const float* __restrict__ g, const float* __restrict__ cnt,
                           const float* __restrict__ w1, const float* __restrict__ b1,
                           const float* __restrict__ w2, const float* __restrict__ b2,
                           float* __restrict__ out)
{
  __shared__ float gv[96], h[96];
  int b = blockIdx.x, t = threadIdx.x;
  float c = cnt[b]; c = c < 1.f ? 1.f : c;
  if (t < 96) gv[t] = g[b*96 + t] / c;
  __syncthreads();
  if (t < 96) {
    float a = b1[t];
    for (int i = 0; i < 96; ++i) a += gv[i]*w1[i*96 + t];
    h[t] = silu_f(a);
  }
  __syncthreads();
  if (t == 0) {
    float o = b2[0];
    for (int j = 0; j < 96; ++j) o += h[j]*w2[j];
    out[b] = o;
  }
}

// ------------------------------- host launch -------------------------------
extern "C" void kernel_launch(void* const* d_in, const int* in_sizes, int n_in,
                              void* d_out, int out_size, void* d_ws, size_t ws_size,
                              hipStream_t stream)
{
  (void)in_sizes; (void)n_in; (void)out_size; (void)ws_size;
  const float* x          = (const float*)d_in[0];
  const int*   edge_index = (const int*)  d_in[1];
  const float* edge_attr  = (const float*)d_in[2];
  const float* node_attr  = (const float*)d_in[3];
  const int*   batch      = (const int*)  d_in[4];
  auto Pf = [&](int i){ return (const float*)d_in[i]; };

  // Assumed flattening: setup_inputs() insertion order; each tp-param dict in
  // make_tp_params insertion order (w000,w110,w011,w101,w111,bias).
  const int EMB = 5;
  const int L0 = 11, LSTRIDE = 24;
  const int PRE1 = 59, PRE2 = 65;
  const int POST1W = 71, POST1B = 72, POST2W = 73, POST2B = 74;

  // workspace carve: packed bf16 weights first (uints), then f32 state
  unsigned* wpk = (unsigned*)d_ws;
  const int O_M1S=0, O_M1V=9216, O_M2S=13312, O_M2V=17920,
            O_U1S=19968, O_U1V=29184, O_U2S=33280, O_U2V=36352, O_LST=38400;
  const int O_P1S=76800, O_P1V=81408, O_P2S=83456, O_END=88064;
  float* fb   = (float*)((char*)d_ws + (size_t)O_END*4);
  float* sbuf = fb;
  float* vbuf = sbuf + (size_t)N_NODES*64;
  float* agS  = vbuf + (size_t)N_NODES*96;
  float* agV  = agS  + (size_t)N_NODES*64;
  float* psb  = agV  + (size_t)N_NODES*96;
  float* gb   = psb  + (size_t)N_NODES*96;
  float* cntb = gb   + (size_t)N_GRAPHS*96;

  auto packS = [&](int base, int n0, int n1, int m, int off){
    int tot = ((n0+n1)/2)*m;
    pack_stacked<<<(tot+255)/256, 256, 0, stream>>>(
        Pf(base+0), n0, 1.f, Pf(base+1), n1, INV3F, nullptr, 0, 0.f, m, wpk+off);
  };
  auto packV = [&](int base, int n0, int n1, int m, int off){
    int tot = ((n0+2*n1)/2)*m;
    pack_stacked<<<(tot+255)/256, 256, 0, stream>>>(
        Pf(base+2), n0, 1.f, Pf(base+3), n1, 1.f, Pf(base+4), n1, INV2F, m, wpk+off);
  };
  for (int l = 0; l < 2; ++l) {
    int L = L0 + l*LSTRIDE, O = l*O_LST;
    packS(L+0, 128, 64, 96, O+O_M1S);  packV(L+0, 128, 64, 32, O+O_M1V);
    packS(L+6,  64, 32, 96, O+O_M2S);  packV(L+6,  64, 32, 32, O+O_M2V);
    packS(L+12,128, 64, 96, O+O_U1S);  packV(L+12,128, 64, 32, O+O_U1V);
    packS(L+18, 64, 32, 64, O+O_U2S);  packV(L+18, 64, 32, 32, O+O_U2V);
  }
  packS(PRE1, 64, 32, 96, O_P1S);  packV(PRE1, 64, 32, 32, O_P1V);
  packS(PRE2, 64, 32, 96, O_P2S);

  embed_kernel<<<(N_NODES+255)/256, 256, 0, stream>>>(
      x, node_attr, Pf(EMB+0), Pf(EMB+1), Pf(EMB+2), Pf(EMB+3), Pf(EMB+4), Pf(EMB+5),
      sbuf, vbuf);

  hipFuncSetAttribute(reinterpret_cast<const void*>(message_kernel),
                      hipFuncAttributeMaxDynamicSharedMemorySize, SMEM_MSG);
  hipFuncSetAttribute(reinterpret_cast<const void*>(update_kernel),
                      hipFuncAttributeMaxDynamicSharedMemorySize, SMEM_UPD);
  hipFuncSetAttribute(reinterpret_cast<const void*>(pre_kernel),
                      hipFuncAttributeMaxDynamicSharedMemorySize, SMEM_PRE);

  for (int l = 0; l < 2; ++l) {
    int L = L0 + l*LSTRIDE, O = l*O_LST;
    zero_kernel<<<4096, 256, 0, stream>>>(agS, (size_t)N_NODES*(64+96)); // agS+agV contiguous
    message_kernel<<<N_EDGES/64, 128, SMEM_MSG, stream>>>(
        sbuf, vbuf, edge_index, edge_attr,
        wpk+O+O_M1S, wpk+O+O_M1V, Pf(L+5),
        wpk+O+O_M2S, wpk+O+O_M2V, Pf(L+11),
        agS, agV);
    update_kernel<<<(N_NODES+63)/64, 128, SMEM_UPD, stream>>>(
        sbuf, vbuf, agS, agV, node_attr,
        wpk+O+O_U1S, wpk+O+O_U1V, Pf(L+17),
        wpk+O+O_U2S, wpk+O+O_U2V, Pf(L+23));
  }

  pre_kernel<<<(N_NODES+63)/64, 128, SMEM_PRE, stream>>>(
      sbuf, vbuf, node_attr,
      wpk+O_P1S, wpk+O_P1V, Pf(PRE1+5),
      wpk+O_P2S, Pf(PRE2+5), psb);

  zero_kernel<<<64, 256, 0, stream>>>(gb, (size_t)N_GRAPHS*96 + N_GRAPHS); // g + cnt
  pool_kernel<<<(N_NODES*96+255)/256, 256, 0, stream>>>(psb, batch, gb, cntb);
  mlp_kernel<<<N_GRAPHS, 128, 0, stream>>>(gb, cntb, Pf(POST1W), Pf(POST1B),
                                           Pf(POST2W), Pf(POST2B), (float*)d_out);
}
